// VectorQuantizer_47545287967483
// MI455X (gfx1250) — matrix operand order repaired and checked by reading, not compiled, not run on hardware
//
#include <hip/hip_runtime.h>
#include <hip/hip_bf16.h>
#include <math.h>

// ---------------------------------------------------------------------------
// VQ (8 codebooks x 512 codes x dim 32) on gfx1250 via v_wmma_f32_16x16x32_bf16
//   score(code, tok) = 16 + ||cb||^2 - 2*cb.x   ( = dist - ||x||^2 + 16 > 0 )
// A = (-2*cb) 16x32 bf16 tile from LDS, B = x 32x16 bf16 tile, C = 16+||cb||^2.
// Argmin via packed u32 keys: (f32 bits & ~0x1FF) | code_index, v_min3_u32 trees.
// Grid split over codebooks: block = (codebook c, 512-token group); 34KB LDS
// per block so multiple blocks co-reside per WGP to hide LDS/HBM latency.
// Branch-free software pipeline: k-loop prefetches A/C of k16+1, last WMMA peeled.
// ---------------------------------------------------------------------------

typedef __attribute__((ext_vector_type(16))) __bf16         v16bf;
typedef __attribute__((ext_vector_type(8)))  float          v8f;
typedef __attribute__((ext_vector_type(16))) unsigned short v16u;
typedef __attribute__((ext_vector_type(8)))  unsigned short v8u;

#define NUM_CB        8
#define CB_SIZE       512
#define SUB_D         32
#define D_TOT         256
#define NTOK          65536                    // 16 * 4096 tokens
#define KTILES        (CB_SIZE / 16)           // 32 code tiles
#define TILES_PER_WAVE 4
#define TOK_PER_BLOCK (8 * TILES_PER_WAVE * 16)   // 512 tokens
#define NGROUPS       (NTOK / TOK_PER_BLOCK)      // 128
#define GRID_MAIN     (NGROUPS * NUM_CB)          // 1024 blocks
#define ELEMS_X       16777216.0f              // 16*4096*8*32 (loss denominator)
#define DIST_BIAS     16.0f                    // makes all scores > 0 (|2x.cb| < ~2)

// output layout (floats, concatenated in reference return order)
#define OFF_IDS    0
#define N_IDS      (NTOK * NUM_CB)             // 524288
#define OFF_QUANT  (OFF_IDS + N_IDS)
#define N_FEAT     (NTOK * D_TOT)              // 16777216
#define OFF_ST     (OFF_QUANT + N_FEAT)
#define OFF_SCAL   (OFF_ST + N_FEAT)           // 34078720

__device__ __forceinline__ unsigned short f2bf(float f) {
    unsigned int u = __builtin_bit_cast(unsigned int, f);
    unsigned int r = (u + 0x7FFFu + ((u >> 16) & 1u)) >> 16;   // round-nearest-even
    return (unsigned short)r;
}

__device__ __forceinline__ unsigned int umin32(unsigned int a, unsigned int b) {
    return a < b ? a : b;
}

__global__ void vq_zero(float* __restrict__ ws) {
    int i = blockIdx.x * blockDim.x + threadIdx.x;
    if (i < 2 + NUM_CB * CB_SIZE) ws[i] = 0.f;
}

__global__ __launch_bounds__(256)
void vq_main(const float* __restrict__ latents,
             const float* __restrict__ pmask,
             const float* __restrict__ codebooks,
             float* __restrict__ outIds,
             float* __restrict__ outQuant,
             float* __restrict__ outSt,
             float* __restrict__ ws)
{
    __shared__ unsigned short cbA[CB_SIZE * SUB_D];  // 32KB: bf16(-2*cb), [code][dim]
    __shared__ float          cbn[CB_SIZE];          //  2KB: DIST_BIAS + ||cb||^2
    __shared__ float          red[16];               // block reduction scratch

    const int tid  = threadIdx.x;
    const int lane = tid & 31;
    const int wid  = tid >> 5;                 // 8 waves, each owns token tiles
    const int tl   = lane & 15;
    const int hh   = lane >> 4;                // lane half (WMMA K / M split)
    const int c    = blockIdx.x & (NUM_CB - 1);
    const int grp  = blockIdx.x >> 3;

    // ---- Stage this block's codebook into LDS: bf16(-2*cb) + biased norms ----
    for (int code = tid; code < CB_SIZE; code += 256) {
        const float4*   src = (const float4*)(codebooks + (size_t)(c * CB_SIZE + code) * SUB_D);
        unsigned short* dst = cbA + code * SUB_D;
        float nrm = DIST_BIAS;
        #pragma unroll
        for (int q = 0; q < 8; ++q) {
            float4 v = src[q];
            nrm += v.x*v.x + v.y*v.y + v.z*v.z + v.w*v.w;
            dst[q*4+0] = f2bf(-2.f * v.x);
            dst[q*4+1] = f2bf(-2.f * v.y);
            dst[q*4+2] = f2bf(-2.f * v.z);
            dst[q*4+3] = f2bf(-2.f * v.w);
        }
        cbn[code] = nrm;
    }
    __syncthreads();

    auto loadA = [&](int k16, v16bf& af, v8f& cf) {
        // A fragment: lane holds code row M = tl; elements 0..15 = dims hh*16..+15
        // (contiguous K order, matching B)
        const int codeA = k16 * 16 + tl;
        const v8u* ap = (const v8u*)(cbA + codeA * SUB_D + hh * 16);
        v8u lo = ap[0];
        v8u hi = ap[1];
        v16u au;
        #pragma unroll
        for (int i = 0; i < 8; ++i) { au[i] = lo[i]; au[8+i] = hi[i]; }
        af = __builtin_bit_cast(v16bf, au);
        // C fragment: biased norm for output code row m = k16*16 + 8*hh + j
        const int mb = k16 * 16 + 8 * hh;
        #pragma unroll
        for (int j = 0; j < 8; ++j) cf[j] = cbn[mb + j];
    };

    float local_sse = 0.f;
    float local_msk = 0.f;

    for (int tt = 0; tt < TILES_PER_WAVE; ++tt) {
        const int tok = grp * TOK_PER_BLOCK + (wid * TILES_PER_WAVE + tt) * 16 + tl;

        // B fragment (32x16 bf16): lanes 0-15 dims 0..15, lanes 16-31 dims 16..31
        const float4* xp = (const float4*)(latents + (size_t)tok * D_TOT + c * SUB_D + hh * 16);
        v16u bu;
        #pragma unroll
        for (int q = 0; q < 4; ++q) {
            float4 v = xp[q];
            bu[q*4+0] = f2bf(v.x);
            bu[q*4+1] = f2bf(v.y);
            bu[q*4+2] = f2bf(v.z);
            bu[q*4+3] = f2bf(v.w);
        }
        v16bf bfrag = __builtin_bit_cast(v16bf, bu);

        unsigned int bestk = 0xFFFFFFFFu;

        // pack (positive) scores with 9-bit code index; u32-min == argmin
        auto reduceTile = [&](const v8f& dm, int k16) {
            const unsigned int ib = (unsigned int)(k16 * 16 + 8 * hh);
            unsigned int k0 = (__builtin_bit_cast(unsigned int, dm[0]) & 0xFFFFFE00u) | (ib + 0);
            unsigned int k1 = (__builtin_bit_cast(unsigned int, dm[1]) & 0xFFFFFE00u) | (ib + 1);
            unsigned int k2 = (__builtin_bit_cast(unsigned int, dm[2]) & 0xFFFFFE00u) | (ib + 2);
            unsigned int k3 = (__builtin_bit_cast(unsigned int, dm[3]) & 0xFFFFFE00u) | (ib + 3);
            unsigned int k4 = (__builtin_bit_cast(unsigned int, dm[4]) & 0xFFFFFE00u) | (ib + 4);
            unsigned int k5 = (__builtin_bit_cast(unsigned int, dm[5]) & 0xFFFFFE00u) | (ib + 5);
            unsigned int k6 = (__builtin_bit_cast(unsigned int, dm[6]) & 0xFFFFFE00u) | (ib + 6);
            unsigned int k7 = (__builtin_bit_cast(unsigned int, dm[7]) & 0xFFFFFE00u) | (ib + 7);
            unsigned int m0 = umin32(umin32(k0, k1), k2);     // v_min3_u32
            unsigned int m1 = umin32(umin32(k3, k4), k5);
            unsigned int m2 = umin32(umin32(k6, k7), bestk);
            bestk = umin32(umin32(m0, m1), m2);
        };

        // branch-free software pipeline over the 32 code tiles
        v16bf acur; v8f cfcur;
        loadA(0, acur, cfcur);
        for (int k16 = 0; k16 < KTILES - 1; ++k16) {
            v16bf anxt; v8f cfnxt;
            loadA(k16 + 1, anxt, cfnxt);
            v8f dm = __builtin_amdgcn_wmma_f32_16x16x32_bf16(
                         false, acur, false, bfrag, (short)0, cfcur, false, false);
            reduceTile(dm, k16);
            acur = anxt; cfcur = cfnxt;
        }
        {
            v8f dm = __builtin_amdgcn_wmma_f32_16x16x32_bf16(
                         false, acur, false, bfrag, (short)0, cfcur, false, false);
            reduceTile(dm, KTILES - 1);
        }

        // merge lane halves (codes m and m+8 live in lane and lane^16)
        unsigned int other = (unsigned int)__shfl_xor((int)bestk, 16);
        bestk = umin32(bestk, other);
        const int besti = (int)(bestk & 511u);

        if (hh == 0) {
            outIds[(size_t)tok * NUM_CB + c] = (float)besti;
            const float4* cq = (const float4*)(codebooks + ((size_t)(c * CB_SIZE + besti)) * SUB_D);
            const float4* xx = (const float4*)(latents + (size_t)tok * D_TOT + c * SUB_D);
            float4* q1 = (float4*)(outQuant + (size_t)tok * D_TOT + c * SUB_D);
            float4* q2 = (float4*)(outSt   + (size_t)tok * D_TOT + c * SUB_D);
            #pragma unroll
            for (int q = 0; q < 8; ++q) {
                float4 cv = cq[q];
                float4 xv = xx[q];
                q1[q] = cv;           // quant
                q2[q] = cv;           // st_quantized == quant numerically
                float dx = xv.x - cv.x, dy = xv.y - cv.y;
                float dz = xv.z - cv.z, dw = xv.w - cv.w;
                local_sse += dx*dx + dy*dy + dz*dz + dw*dw;
            }
            float m = pmask[tok];
            atomicAdd(&ws[2 + c * CB_SIZE + besti], m);   // usage histogram
            if (c == 0) local_msk += m;
        }
    }

    // block reduction: SSE + mask-sum, one global atomic per block
    float s = local_sse;
    #pragma unroll
    for (int off = 16; off > 0; off >>= 1) s += __shfl_down(s, off);
    float mm = local_msk;
    #pragma unroll
    for (int off = 16; off > 0; off >>= 1) mm += __shfl_down(mm, off);
    if (lane == 0) { red[wid] = s; red[8 + wid] = mm; }
    __syncthreads();
    if (tid == 0) {
        float ts = 0.f, tm = 0.f;
        #pragma unroll
        for (int w = 0; w < 8; ++w) { ts += red[w]; tm += red[8 + w]; }
        atomicAdd(&ws[0], ts);
        if (c == 0) atomicAdd(&ws[1], tm);
    }
}

__global__ void vq_finalize(const float* __restrict__ ws, float* __restrict__ outScal) {
    __shared__ float Hc[NUM_CB];
    const int tid = threadIdx.x;
    if (tid < NUM_CB) Hc[tid] = 0.f;
    __syncthreads();
    const float denom = fmaxf(ws[1], 1.0f);
    for (int i = tid; i < NUM_CB * CB_SIZE; i += 256) {
        float p = ws[2 + i] / denom;
        atomicAdd(&Hc[i >> 9], -(p * logf(p + 1e-8f)));
    }
    __syncthreads();
    if (tid == 0) {
        float mse = ws[0] / ELEMS_X;
        outScal[0] = 0.25f * mse;   // commitment_loss
        outScal[1] = 1.00f * mse;   // codebook_loss
        float perp = 0.f;
        #pragma unroll
        for (int cc = 0; cc < NUM_CB; ++cc) perp += expf(Hc[cc]);
        outScal[2] = perp / (float)NUM_CB;
    }
}

extern "C" void kernel_launch(void* const* d_in, const int* in_sizes, int n_in,
                              void* d_out, int out_size, void* d_ws, size_t ws_size,
                              hipStream_t stream) {
    const float* latents   = (const float*)d_in[0];   // (16,4096,256) f32
    const float* pmask     = (const float*)d_in[1];   // (16,4096)     f32
    const float* codebooks = (const float*)d_in[2];   // (8,512,32)    f32
    float* out = (float*)d_out;
    float* ws  = (float*)d_ws;

    vq_zero<<<(2 + NUM_CB * CB_SIZE + 255) / 256, 256, 0, stream>>>(ws);
    vq_main<<<GRID_MAIN, 256, 0, stream>>>(
        latents, pmask, codebooks,
        out + OFF_IDS, out + OFF_QUANT, out + OFF_ST, ws);
    vq_finalize<<<1, 256, 0, stream>>>(ws, out + OFF_SCAL);
}
